// TransMIL_LinearPE_End_23536420782301
// MI455X (gfx1250) — matrix operand order repaired and checked by reading, not compiled, not run on hardware
//
#include <hip/hip_runtime.h>

typedef long long i64;
typedef __bf16 bf16;
typedef __attribute__((ext_vector_type(16))) __bf16 v16bf;
typedef __attribute__((ext_vector_type(8)))  __bf16 v8bf;
typedef __attribute__((ext_vector_type(8)))  float  v8f;

// ---------------- model constants ----------------
constexpr int B_ = 2;
constexpr int N_ = 6000;        // raw tokens
constexpr int OUTER_ = 1024;
constexpr int DIM_ = 512;
constexpr int T_ = 6144;        // padded sequence frame: 59 zero-pad + CLS + 6084 tokens
constexpr int PADF_ = 59;       // front pad
constexpr int H_ = 8;
constexpr int DH_ = 64;
constexpr int LM_ = 256;        // landmarks
constexpr int LPB_ = 24;        // tokens per landmark (6144/256)
constexpr float SCALE_ = 0.125f; // 64^-0.5

// ---------------- reductions ----------------
__device__ inline float wsum(float v) {
#pragma unroll
  for (int o = 16; o > 0; o >>= 1) v += __shfl_xor(v, o, 32);
  return v;
}
__device__ inline float wmaxr(float v) {
#pragma unroll
  for (int o = 16; o > 0; o >>= 1) v = fmaxf(v, __shfl_xor(v, o, 32));
  return v;
}
__device__ inline float blockSum(float v, float* sbuf, int nw) {
  v = wsum(v);
  int w = threadIdx.x >> 5, l = threadIdx.x & 31;
  __syncthreads();
  if (l == 0) sbuf[w] = v;
  __syncthreads();
  float r = (l < nw) ? sbuf[l] : 0.f;
  return wsum(r);
}
__device__ inline float blockMax(float v, float* sbuf, int nw) {
  v = wmaxr(v);
  int w = threadIdx.x >> 5, l = threadIdx.x & 31;
  __syncthreads();
  if (l == 0) sbuf[w] = v;
  __syncthreads();
  float r = (l < nw) ? sbuf[l] : -3.4e38f;
  return wmaxr(r);
}

// ---------------- WMMA fragment loaders (ISA 7.12.2 layouts) ----------------
// A 16x32 bf16: lane=16h+m holds row m, K in [8h,8h+8) and [16+8h,16+8h+8)
__device__ inline v16bf ldfragA(const bf16* row, int h) {
  v8bf lo = *(const v8bf*)(row + 8 * h);
  v8bf hi = *(const v8bf*)(row + 16 + 8 * h);
  v16bf r;
#pragma unroll
  for (int i = 0; i < 8; ++i) { r[i] = lo[i]; r[i + 8] = hi[i]; }
  return r;
}
// B 32x16 bf16 (stored as B^T rows, [N][K]): lane=16h+n holds col n, K in [8h,8h+8) and [16+8h,16+8h+8)
__device__ inline v16bf ldfragB(const bf16* row, int h) {
  v8bf lo = *(const v8bf*)(row + 8 * h);
  v8bf hi = *(const v8bf*)(row + 16 + 8 * h);
  v16bf r;
#pragma unroll
  for (int i = 0; i < 8; ++i) { r[i] = lo[i]; r[i + 8] = hi[i]; }
  return r;
}

// ---------------- generic batched WMMA GEMM ----------------
// C[M,N] = A[M,K] * op(B),  op(B)=B^T if TRANSB (B stored [N,K]), else B ([K,N]).
// Block tile 256x64, BK=32, 8 waves; each wave computes a 32x64 tile (2x4 WMMA
// subtiles, A-frags reused across 4 n-tiles, B-frags reused across 2 m-subtiles).
// Batch z: offset = (z/zdiv)*Outer + (z%zdiv)*Inner per operand (handles [b,h] splits).
constexpr int EPI_PLAIN = 0, EPI_BIAS_RELU = 1, EPI_BF16 = 2, EPI_BOTH = 3, EPI_RESID = 4;

template <typename T> struct __attribute__((aligned(4 * sizeof(T)))) V4 { T a, b, c, d; };

template <typename TA, typename TB, bool TRANSB, int EPI>
__global__ __launch_bounds__(256) void gemm_k(
    const TA* __restrict__ A, i64 lda, i64 aOuter, i64 aInner,
    const TB* __restrict__ Bm, i64 ldb, i64 bOuter, i64 bInner,
    float* __restrict__ C, bf16* __restrict__ Cb,
    i64 ldc, i64 cOuter, i64 cInner,
    const float* __restrict__ bias, float alpha,
    int M, int N, int K, int zdiv, int Tres, int tmin) {
  __shared__ bf16 As[256 * 32];   // 16 KB
  __shared__ bf16 Bs[64 * 32];    //  4 KB
  const int tid = threadIdx.x;
  const int wave = tid >> 5, lane = tid & 31;
  const int hh = lane >> 4, ln = lane & 15;
  const i64 z = blockIdx.z;
  const i64 zo = z / zdiv, zi = z % zdiv;
  const TA* Ab = A + zo * aOuter + zi * aInner;
  const TB* Bb = Bm + zo * bOuter + zi * bInner;
  const i64 coff = zo * cOuter + zi * cInner;
  const int m0 = blockIdx.y * 256;
  const int n0 = blockIdx.x * 64;

  v8f acc[2][4];
#pragma unroll
  for (int mi = 0; mi < 2; ++mi)
#pragma unroll
    for (int t = 0; t < 4; ++t)
#pragma unroll
      for (int i = 0; i < 8; ++i) acc[mi][t][i] = 0.f;

  for (int k0 = 0; k0 < K; k0 += 32) {
    // prefetch next A k-tile (hint only; lowers to global_prefetch_b8)
    if (k0 + 32 < K) {
      int r = tid >> 3, cg = (tid & 7) * 4;
      int gr = m0 + r;
      if (gr < M)
        __builtin_prefetch((const void*)(Ab + (i64)gr * lda + k0 + 32 + cg), 0, 1);
    }
    // stage A tile (256x32), branchless edge guard, f32/bf16 -> bf16 convert
#pragma unroll
    for (int i = 0; i < 8; ++i) {
      int e = tid + i * 256;            // 0..2047, 4 elems each
      int r = e >> 3, cg = (e & 7) * 4;
      int gr = m0 + r;
      bool inr = (gr < M);
      i64 ro = (i64)(inr ? gr : 0) * lda + k0 + cg;
      V4<TA> q = *(const V4<TA>*)(Ab + ro);
      bf16* d = &As[r * 32 + cg];
      d[0] = inr ? (bf16)(float)q.a : (bf16)0.f;
      d[1] = inr ? (bf16)(float)q.b : (bf16)0.f;
      d[2] = inr ? (bf16)(float)q.c : (bf16)0.f;
      d[3] = inr ? (bf16)(float)q.d : (bf16)0.f;
    }
    // stage B tile as [n][k] (64x32)
    if (TRANSB) {
#pragma unroll
      for (int i = 0; i < 2; ++i) {
        int e = tid + i * 256;          // 0..511
        int r = e >> 3, cg = (e & 7) * 4;
        V4<TB> q = *(const V4<TB>*)(Bb + (i64)(n0 + r) * ldb + k0 + cg);
        bf16* d = &Bs[r * 32 + cg];
        d[0] = (bf16)(float)q.a; d[1] = (bf16)(float)q.b;
        d[2] = (bf16)(float)q.c; d[3] = (bf16)(float)q.d;
      }
    } else {
#pragma unroll
      for (int i = 0; i < 2; ++i) {
        int e = tid + i * 256;          // 0..511
        int kk = e >> 4, n4 = (e & 15) * 4;
        V4<TB> q = *(const V4<TB>*)(Bb + (i64)(k0 + kk) * ldb + n0 + n4);
        Bs[(n4 + 0) * 32 + kk] = (bf16)(float)q.a;
        Bs[(n4 + 1) * 32 + kk] = (bf16)(float)q.b;
        Bs[(n4 + 2) * 32 + kk] = (bf16)(float)q.c;
        Bs[(n4 + 3) * 32 + kk] = (bf16)(float)q.d;
      }
    }
    __syncthreads();
    v16bf af0 = ldfragA(&As[(wave * 32 + ln) * 32], hh);
    v16bf af1 = ldfragA(&As[(wave * 32 + 16 + ln) * 32], hh);
#pragma unroll
    for (int nt = 0; nt < 4; ++nt) {
      v16bf bf = ldfragB(&Bs[(nt * 16 + ln) * 32], hh);
      acc[0][nt] = __builtin_amdgcn_wmma_f32_16x16x32_bf16(
          false, af0, false, bf, (short)0, acc[0][nt], false, false);
      acc[1][nt] = __builtin_amdgcn_wmma_f32_16x16x32_bf16(
          false, af1, false, bf, (short)0, acc[1][nt], false, false);
    }
    __syncthreads();
  }
  // epilogue: C layout lane=16h+n holds col n, VGPR r -> row r+8h
#pragma unroll
  for (int mi = 0; mi < 2; ++mi) {
#pragma unroll
    for (int nt = 0; nt < 4; ++nt) {
      int col = n0 + nt * 16 + ln;
#pragma unroll
      for (int r = 0; r < 8; ++r) {
        int row = m0 + wave * 32 + mi * 16 + hh * 8 + r;
        if (row >= M) continue;
        float v = acc[mi][nt][r];
        i64 o = coff + (i64)row * ldc + col;
        if constexpr (EPI == EPI_PLAIN) {
          C[o] = v;
        } else if constexpr (EPI == EPI_BIAS_RELU) {
          v += bias[col]; C[o] = fmaxf(v, 0.f);
        } else if constexpr (EPI == EPI_BF16) {
          Cb[o] = (bf16)v;
        } else if constexpr (EPI == EPI_BOTH) {
          v *= alpha; if (C) C[o] = v; Cb[o] = (bf16)v;
        } else {  // EPI_RESID: h += out + bias, skipping front-pad rows
          v += bias[col];
          if ((row % Tres) >= tmin) C[o] += v;
        }
      }
    }
  }
}

// ---------------- small kernels ----------------
// zero pad rows + write CLS row of h. grid = B*(PADF_+1), block 512
__global__ __launch_bounds__(512) void init_h_k(float* h, const float* cls) {
  int id = blockIdx.x;
  int b = id / (PADF_ + 1), t = id % (PADF_ + 1);
  float* row = h + ((i64)b * T_ + t) * DIM_;
  row[threadIdx.x] = (t == PADF_) ? cls[threadIdx.x] : 0.f;
}

// copy wrapped square-padding rows: token 6000+j = token j. grid = B*84, block 512
__global__ __launch_bounds__(512) void wrap_h_k(float* h) {
  int id = blockIdx.x;
  int b = id / 84, j = id % 84;
  h[((i64)b * T_ + (PADF_ + 1) + 6000 + j) * DIM_ + threadIdx.x] =
      h[((i64)b * T_ + (PADF_ + 1) + j) * DIM_ + threadIdx.x];
}

// per-token LayerNorm -> bf16; pad rows -> 0. grid = B*T_, block 256
__global__ __launch_bounds__(256) void ln_rows_k(const float* h, const float* g,
                                                 const float* bb, bf16* out) {
  __shared__ float sbuf[8];
  i64 idx = blockIdx.x;
  int t = (int)(idx % T_);
  int tid = threadIdx.x;
  bf16* o = out + idx * DIM_;
  if (t < PADF_) { o[tid] = (bf16)0.f; o[tid + 256] = (bf16)0.f; return; }
  const float* x = h + idx * DIM_;
  float a = x[tid], b2 = x[tid + 256];
  float s = blockSum(a + b2, sbuf, 8);
  float s2 = blockSum(a * a + b2 * b2, sbuf, 8);
  float mu = s * (1.f / DIM_);
  float var = s2 * (1.f / DIM_) - mu * mu;
  float rstd = rsqrtf(var + 1e-5f);
  o[tid] = (bf16)((a - mu) * rstd * g[tid] + bb[tid]);
  o[tid + 256] = (bf16)((b2 - mu) * rstd * g[tid + 256] + bb[tid + 256]);
}

// landmark means: ql (scaled), kl. grid = 16*LM_ (bh,j), block 64 (d)
__global__ __launch_bounds__(64) void landmarks_k(const bf16* qkv, bf16* ql, bf16* kl) {
  int id = blockIdx.x;
  int bh = id >> 8, j = id & 255;
  int b = bh >> 3, hh = bh & 7;
  int d = threadIdx.x;
  i64 base = ((i64)(b * T_ + j * LPB_)) * 1536 + hh * 64 + d;
  float qs = 0.f, ks = 0.f;
#pragma unroll 4
  for (int i = 0; i < LPB_; ++i) {
    qs += (float)qkv[base + (i64)i * 1536];
    ks += (float)qkv[base + (i64)i * 1536 + 512];
  }
  ql[(i64)id * 64 + d] = (bf16)(qs * (SCALE_ / LPB_));
  kl[(i64)id * 64 + d] = (bf16)(ks * (1.f / LPB_));
}

// softmax over rows; logits = scale*in. grid = rows, block 256
__global__ __launch_bounds__(256) void softmax_rows_k(const float* in, bf16* ob,
                                                      float* of, int cols, float scale) {
  __shared__ float sbuf[8];
  i64 row = blockIdx.x;
  const float* p = in + row * (i64)cols;
  float m = -3.4e38f;
  for (int c = threadIdx.x; c < cols; c += 256) m = fmaxf(m, p[c]);
  m = blockMax(m, sbuf, 8);
  float s = 0.f;
  for (int c = threadIdx.x; c < cols; c += 256) s += __expf(scale * (p[c] - m));
  s = blockSum(s, sbuf, 8);
  float inv = 1.f / s;
  for (int c = threadIdx.x; c < cols; c += 256) {
    float v = __expf(scale * (p[c] - m)) * inv;
    ob[row * (i64)cols + c] = (bf16)v;
    if (of) of[row * (i64)cols + c] = v;
  }
}

// abs col/row sums of attn2 (x[16,256,256])
__global__ __launch_bounds__(256) void colsum_k(const float* x, float* cs) {
  __shared__ float sbuf[8];
  i64 id = blockIdx.x;  // bh*256 + i
  float v = fabsf(x[id * 256 + threadIdx.x]);
  v = blockSum(v, sbuf, 8);
  if (threadIdx.x == 0) cs[id] = v;
}
__global__ __launch_bounds__(256) void rowsum_k(const float* x, float* rs) {
  __shared__ float sbuf[8];
  i64 id = blockIdx.x;  // bh*256 + j
  i64 bh = id >> 8; int j = (int)(id & 255);
  float v = fabsf(x[bh * 65536 + (i64)threadIdx.x * 256 + j]);
  v = blockSum(v, sbuf, 8);
  if (threadIdx.x == 0) rs[id] = v;
}
__global__ __launch_bounds__(256) void scalarmax_k(const float* cs, const float* rs,
                                                   float* denom, int n) {
  __shared__ float sbuf[8];
  float m1 = -3.4e38f, m2 = -3.4e38f;
  for (int i = threadIdx.x; i < n; i += 256) { m1 = fmaxf(m1, cs[i]); m2 = fmaxf(m2, rs[i]); }
  m1 = blockMax(m1, sbuf, 8);
  m2 = blockMax(m2, sbuf, 8);
  if (threadIdx.x == 0) denom[0] = 1.f / (m1 * m2);
}
// z0 = x^T / (maxcol*maxrow). grid = 16*256 (bh,i), block 256 (j)
__global__ __launch_bounds__(256) void zinit_k(const float* x, const float* denom, bf16* z) {
  i64 id = blockIdx.x;
  i64 bh = id >> 8; int i = (int)(id & 255), j = threadIdx.x;
  z[bh * 65536 + (i64)i * 256 + j] = (bf16)(x[bh * 65536 + (i64)j * 256 + i] * denom[0]);
}
// out = bf16(alpha*I - in), over 16*256*256
__global__ __launch_bounds__(256) void ew_aim_k(const float* a, bf16* o, float alpha) {
  i64 idx = (i64)blockIdx.x * 256 + threadIdx.x;
  int w = (int)(idx & 65535);
  int i = w >> 8, j = w & 255;
  o[idx] = (bf16)(((i == j) ? alpha : 0.f) - a[idx]);
}

// depthwise 33-tap conv residual over sequence + head interleave -> bf16 [B,T,512]
// grid = B*T_, block 512 (h*64+d)
__global__ __launch_bounds__(512) void conv_reorder_k(const float* out1c, const bf16* qkv,
                                                      const float* convw, bf16* outpre) {
  int bt = blockIdx.x;
  int b = bt / T_, t = bt % T_;
  int tid = threadIdx.x;
  int hh = tid >> 6, d = tid & 63;
  float acc = out1c[((i64)(b * 8 + hh) * T_ + t) * 64 + d];
  i64 vbase = (i64)b * T_ * 1536 + 1024 + hh * 64 + d;
#pragma unroll
  for (int k = 0; k < 33; ++k) {
    int tt = t + k - 16;
    if (tt >= 0 && tt < T_)
      acc += convw[hh * 33 + k] * (float)qkv[vbase + (i64)tt * 1536];
  }
  outpre[(i64)bt * 512 + tid] = (bf16)acc;
}

// final LN of CLS token + 2-class head. grid = B_, block 512
__global__ __launch_bounds__(512) void final_k(const float* h, const float* g,
                                               const float* bb, const float* w2,
                                               const float* b2, float* out) {
  __shared__ float xn[512];
  __shared__ float sbuf[16];
  int b = blockIdx.x, tid = threadIdx.x;
  const float* x = h + ((i64)b * T_ + PADF_) * DIM_;
  float v = x[tid];
  float s = blockSum(v, sbuf, 16);
  float s2 = blockSum(v * v, sbuf, 16);
  float mu = s * (1.f / 512.f);
  float var = s2 * (1.f / 512.f) - mu * mu;
  float rstd = rsqrtf(var + 1e-5f);
  xn[tid] = (v - mu) * rstd * g[tid] + bb[tid];
  __syncthreads();
  for (int c = 0; c < 2; ++c) {
    float p = w2[c * 512 + tid] * xn[tid];
    float t = blockSum(p, sbuf, 16);
    if (tid == 0) out[b * 2 + c] = t + b2[c];
  }
}

// ---------------- host-side orchestration ----------------
struct WS {
  float *h, *scores, *out1c, *att2f, *abuf, *cbuf, *colsum, *rowsum, *denom;
  bf16 *xln, *qkv, *attnP, *outpre, *ql, *kl, *att2b, *abf, *tbf, *zbf0, *zbf1, *a3v, *wmat;
};

static void run_layer(const WS& w, const float* lng, const float* lnb, const float* wqkv,
                      const float* wout, const float* bout, const float* convw,
                      hipStream_t s) {
  const i64 QL = (i64)LM_ * DH_;     // 16384
  const i64 MM = (i64)LM_ * LM_;     // 65536

  // x = LayerNorm(h) (pad rows -> 0)
  ln_rows_k<<<B_ * T_, 256, 0, s>>>(w.h, lng, lnb, w.xln);

  // qkv = x @ Wqkv^T  (bf16 out)
  gemm_k<bf16, float, true, EPI_BF16><<<dim3(1536 / 64, (B_ * T_) / 256, 1), 256, 0, s>>>(
      w.xln, DIM_, 0, 0, wqkv, DIM_, 0, 0,
      nullptr, w.qkv, 1536, 0, 0, nullptr, 1.f, B_ * T_, 1536, DIM_, 1, 0, 0);

  // landmark means
  landmarks_k<<<16 * LM_, 64, 0, s>>>(w.qkv, w.ql, w.kl);

  // attn3 logits = q_l @ k^T  -> scores[16,256,T]
  gemm_k<bf16, bf16, true, EPI_PLAIN><<<dim3(T_ / 64, 1, 16), 256, 0, s>>>(
      w.ql, DH_, 8 * QL, QL,
      w.qkv + 512, 1536, (i64)T_ * 1536, 64,
      w.scores, nullptr, T_, 8 * (i64)LM_ * T_, (i64)LM_ * T_,
      nullptr, 1.f, LM_, T_, DH_, 8, 0, 0);
  softmax_rows_k<<<16 * LM_, 256, 0, s>>>(w.scores, w.attnP, nullptr, T_, 1.f);

  // attn3v = attn3 @ v -> [16,256,64] bf16
  gemm_k<bf16, bf16, false, EPI_BF16><<<dim3(1, 1, 16), 256, 0, s>>>(
      w.attnP, T_, 8 * (i64)LM_ * T_, (i64)LM_ * T_,
      w.qkv + 1024, 1536, (i64)T_ * 1536, 64,
      nullptr, w.a3v, DH_, 8 * QL, QL,
      nullptr, 1.f, LM_, DH_, T_, 8, 0, 0);

  // attn2 logits = q_l @ k_l^T, softmax (f32 in-place + bf16)
  gemm_k<bf16, bf16, true, EPI_PLAIN><<<dim3(4, 1, 16), 256, 0, s>>>(
      w.ql, DH_, QL, 0, w.kl, DH_, QL, 0,
      w.att2f, nullptr, LM_, MM, 0, nullptr, 1.f, LM_, LM_, DH_, 1, 0, 0);
  softmax_rows_k<<<16 * LM_, 256, 0, s>>>(w.att2f, w.att2b, w.att2f, LM_, 1.f);

  // Moore-Penrose pinv init: z = x^T / (max colsum * max rowsum)
  colsum_k<<<16 * LM_, 256, 0, s>>>(w.att2f, w.colsum);
  rowsum_k<<<16 * LM_, 256, 0, s>>>(w.att2f, w.rowsum);
  scalarmax_k<<<1, 256, 0, s>>>(w.colsum, w.rowsum, w.denom, 16 * LM_);
  zinit_k<<<16 * LM_, 256, 0, s>>>(w.att2f, w.denom, w.zbf0);

  bf16* zi = w.zbf0;
  bf16* zo = w.zbf1;
  for (int it = 0; it < 6; ++it) {
    // a = x @ z  (f32 + bf16)
    gemm_k<bf16, bf16, false, EPI_BOTH><<<dim3(4, 1, 16), 256, 0, s>>>(
        w.att2b, LM_, MM, 0, zi, LM_, MM, 0,
        w.abuf, w.abf, LM_, MM, 0, nullptr, 1.f, LM_, LM_, LM_, 1, 0, 0);
    ew_aim_k<<<(16 * 65536) / 256, 256, 0, s>>>(w.abuf, w.tbf, 7.f);
    // c = a @ (7I - a)
    gemm_k<bf16, bf16, false, EPI_PLAIN><<<dim3(4, 1, 16), 256, 0, s>>>(
        w.abf, LM_, MM, 0, w.tbf, LM_, MM, 0,
        w.cbuf, nullptr, LM_, MM, 0, nullptr, 1.f, LM_, LM_, LM_, 1, 0, 0);
    ew_aim_k<<<(16 * 65536) / 256, 256, 0, s>>>(w.cbuf, w.tbf, 15.f);
    // e = a @ (15I - c)   (reuse cbuf)
    gemm_k<bf16, bf16, false, EPI_PLAIN><<<dim3(4, 1, 16), 256, 0, s>>>(
        w.abf, LM_, MM, 0, w.tbf, LM_, MM, 0,
        w.cbuf, nullptr, LM_, MM, 0, nullptr, 1.f, LM_, LM_, LM_, 1, 0, 0);
    ew_aim_k<<<(16 * 65536) / 256, 256, 0, s>>>(w.cbuf, w.tbf, 13.f);
    // z' = 0.25 * z @ (13I - e)  (double-buffered)
    gemm_k<bf16, bf16, false, EPI_BOTH><<<dim3(4, 1, 16), 256, 0, s>>>(
        zi, LM_, MM, 0, w.tbf, LM_, MM, 0,
        nullptr, zo, LM_, MM, 0, nullptr, 0.25f, LM_, LM_, LM_, 1, 0, 0);
    bf16* t = zi; zi = zo; zo = t;
  }

  // W = pinv @ (attn3 @ v)
  gemm_k<bf16, bf16, false, EPI_BF16><<<dim3(1, 1, 16), 256, 0, s>>>(
      zi, LM_, MM, 0, w.a3v, DH_, QL, 0,
      nullptr, w.wmat, DH_, QL, 0, nullptr, 1.f, LM_, DH_, LM_, 1, 0, 0);

  // attn1 logits = q @ k_l^T -> scores[16,T,256]; softmax folds SCALE
  gemm_k<bf16, bf16, true, EPI_PLAIN><<<dim3(4, T_ / 256, 16), 256, 0, s>>>(
      w.qkv, 1536, (i64)T_ * 1536, 64,
      w.kl, DH_, 8 * QL, QL,
      w.scores, nullptr, LM_, 8 * (i64)T_ * LM_, (i64)T_ * LM_,
      nullptr, 1.f, T_, LM_, DH_, 8, 0, 0);
  softmax_rows_k<<<16 * T_, 256, 0, s>>>(w.scores, w.attnP, nullptr, LM_, SCALE_);

  // out1 = attn1 @ W -> [16,T,64] f32
  gemm_k<bf16, bf16, false, EPI_PLAIN><<<dim3(1, T_ / 256, 16), 256, 0, s>>>(
      w.attnP, LM_, (i64)T_ * LM_, 0, w.wmat, DH_, QL, 0,
      w.out1c, nullptr, DH_, (i64)T_ * DH_, 0, nullptr, 1.f, T_, DH_, LM_, 1, 0, 0);

  // + depthwise conv(v) residual, interleave heads -> bf16 [B,T,512]
  conv_reorder_k<<<B_ * T_, 512, 0, s>>>(w.out1c, w.qkv, convw, w.outpre);

  // h += outpre @ Wout^T + bout  (skip front-pad rows)
  gemm_k<bf16, float, true, EPI_RESID><<<dim3(DIM_ / 64, (B_ * T_) / 256, 1), 256, 0, s>>>(
      w.outpre, DIM_, 0, 0, wout, DIM_, 0, 0,
      w.h, nullptr, DIM_, 0, 0, bout, 1.f, B_ * T_, DIM_, DIM_, 1, T_, PADF_);
}

extern "C" void kernel_launch(void* const* d_in, const int* in_sizes, int n_in,
                              void* d_out, int out_size, void* d_ws, size_t ws_size,
                              hipStream_t stream) {
  const float* data  = (const float*)d_in[0];
  const float* w1    = (const float*)d_in[1];
  const float* b1    = (const float*)d_in[2];
  const float* cls   = (const float*)d_in[3];
  const float* ln1g  = (const float*)d_in[4];
  const float* ln1b  = (const float*)d_in[5];
  const float* qkv1w = (const float*)d_in[6];
  const float* out1w = (const float*)d_in[7];
  const float* out1b = (const float*)d_in[8];
  const float* conv1 = (const float*)d_in[9];
  const float* ln2g  = (const float*)d_in[10];
  const float* ln2b  = (const float*)d_in[11];
  const float* qkv2w = (const float*)d_in[12];
  const float* out2w = (const float*)d_in[13];
  const float* out2b = (const float*)d_in[14];
  const float* conv2 = (const float*)d_in[15];
  const float* lnfg  = (const float*)d_in[16];
  const float* lnfb  = (const float*)d_in[17];
  const float* w2    = (const float*)d_in[18];
  const float* b2    = (const float*)d_in[19];
  float* out = (float*)d_out;

  // ---- workspace bump allocation (deterministic layout each call) ----
  char* base = (char*)d_ws;
  size_t cur = 0;
  auto alloc = [&](size_t n) -> void* {
    void* r = base + cur;
    cur = (cur + n + 255) & ~(size_t)255;
    return r;
  };
  WS w;
  w.h      = (float*)alloc((size_t)B_ * T_ * DIM_ * 4);
  w.xln    = (bf16*) alloc((size_t)B_ * T_ * DIM_ * 2);
  w.qkv    = (bf16*) alloc((size_t)B_ * T_ * 1536 * 2);
  w.scores = (float*)alloc((size_t)16 * T_ * LM_ * 4);
  w.attnP  = (bf16*) alloc((size_t)16 * T_ * LM_ * 2);
  w.out1c  = (float*)alloc((size_t)16 * T_ * DH_ * 4);
  w.outpre = (bf16*) alloc((size_t)B_ * T_ * DIM_ * 2);
  w.ql     = (bf16*) alloc((size_t)16 * LM_ * DH_ * 2);
  w.kl     = (bf16*) alloc((size_t)16 * LM_ * DH_ * 2);
  w.att2f  = (float*)alloc((size_t)16 * LM_ * LM_ * 4);
  w.att2b  = (bf16*) alloc((size_t)16 * LM_ * LM_ * 2);
  w.abuf   = (float*)alloc((size_t)16 * LM_ * LM_ * 4);
  w.abf    = (bf16*) alloc((size_t)16 * LM_ * LM_ * 2);
  w.tbf    = (bf16*) alloc((size_t)16 * LM_ * LM_ * 2);
  w.cbuf   = (float*)alloc((size_t)16 * LM_ * LM_ * 4);
  w.zbf0   = (bf16*) alloc((size_t)16 * LM_ * LM_ * 2);
  w.zbf1   = (bf16*) alloc((size_t)16 * LM_ * LM_ * 2);
  w.a3v    = (bf16*) alloc((size_t)16 * LM_ * DH_ * 2);
  w.wmat   = (bf16*) alloc((size_t)16 * LM_ * DH_ * 2);
  w.colsum = (float*)alloc((size_t)16 * LM_ * 4);
  w.rowsum = (float*)alloc((size_t)16 * LM_ * 4);
  w.denom  = (float*)alloc(256);
  (void)ws_size; (void)in_sizes; (void)n_in; (void)out_size;

  // h: zero pad rows + CLS row
  init_h_k<<<B_ * (PADF_ + 1), 512, 0, stream>>>(w.h, cls);

  // h[tokens 1..6000] = relu(data @ w1^T + b1), rows start at t = 60
  gemm_k<float, float, true, EPI_BIAS_RELU><<<dim3(DIM_ / 64, (N_ + 255) / 256, B_), 256, 0, stream>>>(
      data, OUTER_, (i64)N_ * OUTER_, 0, w1, OUTER_, 0, 0,
      w.h + (i64)(PADF_ + 1) * DIM_, nullptr, DIM_, (i64)T_ * DIM_, 0,
      b1, 1.f, N_, DIM_, OUTER_, 1, 0, 0);
  // square-padding wrap: tokens 6001..6084 copy tokens 1..84
  wrap_h_k<<<B_ * 84, 512, 0, stream>>>(w.h);

  run_layer(w, ln1g, ln1b, qkv1w, out1w, out1b, conv1, stream);
  run_layer(w, ln2g, ln2b, qkv2w, out2w, out2b, conv2, stream);

  final_k<<<B_, 512, 0, stream>>>(w.h, lnfg, lnfb, w2, b2, out);
}
